// CRNN_498216206720
// MI455X (gfx1250) — matrix operand order repaired and checked by reading, not compiled, not run on hardware
//
#include <hip/hip_runtime.h>
#include <hip/hip_bf16.h>

// Fused conv-GRU (h0 == 0  =>  x_next = tanh(conv_o(x)) * sigmoid(conv_u(x)))
// Cell1 (cin=1) on VALU (memory-bound), cells 2/3 + dense on v_wmma_f32_16x16x32_f16.

typedef __attribute__((ext_vector_type(16))) _Float16 v16h;
typedef __attribute__((ext_vector_type(8)))  float    v8f;

union Frag16 { v16h v; uint4 q[2]; };

// Fast transcendental epilogue: v_rcp_f32 + v_exp_f32 (error << f16 quantization
// already present in the WMMA inputs).
__device__ __forceinline__ float sigm_f(float x) {
  return __builtin_amdgcn_rcpf(1.0f + __expf(-x));
}
__device__ __forceinline__ float tanh_f(float x) {
  return __builtin_fmaf(2.0f, __builtin_amdgcn_rcpf(1.0f + __expf(-2.0f * x)), -1.0f);
}

__device__ __forceinline__ v8f wmma16(v16h a, v16h b, v8f c) {
  return __builtin_amdgcn_wmma_f32_16x16x32_f16(false, a, false, b, (short)0, c, false, false);
}

// d_ws packed f16 weight layout (halves):
//   W2: [128][9][32]  (n<64: w2u couts, n>=64: w2o couts; cin = first 32 chans)
//   W3: [32][9][64]   (n<16: w3u, n>=16: w3o; cin = first 64 chans)
//   Wd: [16][160]     (n<9 real, K = c*9+p, zero-padded K>=144, n>=9)
#define W2_HALVES (128 * 9 * 32)
#define W3_HALVES (32 * 9 * 64)
#define WD_HALVES (16 * 160)
#define W3_OFF W2_HALVES
#define WD_OFF (W2_HALVES + W3_HALVES)

__global__ void crnn_prep_weights(const float* __restrict__ wd,
                                  const float* __restrict__ w2u, const float* __restrict__ w2o,
                                  const float* __restrict__ w3u, const float* __restrict__ w3o,
                                  _Float16* __restrict__ ws) {
  const int total = W2_HALVES + W3_HALVES + WD_HALVES;
  for (int i = blockIdx.x * blockDim.x + threadIdx.x; i < total; i += gridDim.x * blockDim.x) {
    if (i < W2_HALVES) {
      int ci = i & 31, t = (i >> 5) % 9, n = i / (9 * 32);
      // w2u/w2o: [64][96][3][3]; only cin 0..31 used (h part is zero)
      float v = (n < 64) ? w2u[(n * 96 + ci) * 9 + t] : w2o[((n - 64) * 96 + ci) * 9 + t];
      ws[i] = (_Float16)v;
    } else if (i < W2_HALVES + W3_HALVES) {
      int k = i - W2_HALVES;
      int ci = k & 63, t = (k >> 6) % 9, n = k / (9 * 64);
      // w3u/w3o: [16][80][3][3]; only cin 0..63 used
      float v = (n < 16) ? w3u[(n * 80 + ci) * 9 + t] : w3o[((n - 16) * 80 + ci) * 9 + t];
      ws[i] = (_Float16)v;
    } else {
      int k = i - W2_HALVES - W3_HALVES;
      int kk = k % 160, n = k / 160;
      float v = (n < 9 && kk < 144) ? wd[n * 144 + kk] : 0.0f;
      ws[i] = (_Float16)v;
    }
  }
}

__global__ __launch_bounds__(256) void crnn_fused(
    const float* __restrict__ inp,
    const float* __restrict__ w1u, const float* __restrict__ b1u,
    const float* __restrict__ w1o, const float* __restrict__ b1o,
    const float* __restrict__ b2u, const float* __restrict__ b2o,
    const float* __restrict__ b3u, const float* __restrict__ b3o,
    const float* __restrict__ bd,
    const _Float16* __restrict__ wsh,
    float* __restrict__ out, int nb) {
  // LDS activation staging (16 batches per block): rows = batch*9 + pixel.
  // Row 144 of x1h/x2h is an always-zero row used by out-of-bounds conv taps,
  // making the A-fragment ds_load unconditional (one v_cndmask on the row id).
  __shared__ __align__(16) _Float16 x1h[145 * 32];  // [row][cin]
  __shared__ __align__(16) _Float16 x2h[145 * 64];  // [row][cin]
  __shared__ __align__(16) _Float16 x3h[16 * 160];  // [batch][c*9+p], K-padded
  __shared__ __align__(16) float    x0f[144];       // staged inputs [batch][pixel]

  const int tid = threadIdx.x;
  const int b0 = blockIdx.x * 16;

  float* __restrict__ outy  = out;
  float* __restrict__ outx1 = out + (size_t)nb * 9;
  float* __restrict__ outx2 = outx1 + (size_t)nb * 288;
  float* __restrict__ outx3 = outx2 + (size_t)nb * 576;

  const _Float16* __restrict__ W2 = wsh;
  const _Float16* __restrict__ W3 = wsh + W3_OFF;
  const _Float16* __restrict__ Wd = wsh + WD_OFF;

  // ---- phase 0: stage inputs to LDS; zero pad regions ----
  if (tid < 144) x0f[tid] = inp[(size_t)b0 * 9 + tid];
  {
    int bi = tid >> 4, ki = tid & 15;
    x3h[bi * 160 + 144 + ki] = (_Float16)0.0f;        // dense-GEMM K padding
  }
  if (tid < 32) x1h[144 * 32 + tid] = (_Float16)0.0f; // zero row (border taps)
  if (tid < 64) x2h[144 * 64 + tid] = (_Float16)0.0f; // zero row (border taps)
  __syncthreads();

  // ---- phase 1: cell1 (cin=1) on VALU ----
  for (int pair = tid; pair < 16 * 32; pair += 256) {
    int bb = pair >> 5;
    int c  = pair & 31;
    float xv[9];
#pragma unroll
    for (int p = 0; p < 9; p++) xv[p] = x0f[bb * 9 + p];
    float wu_[9], wo_[9];
#pragma unroll
    for (int t = 0; t < 9; t++) {           // w1u/w1o: [32][33][3][3], cin index 0
      wu_[t] = w1u[c * 297 + t];
      wo_[t] = w1o[c * 297 + t];
    }
    const float bu_ = b1u[c], bo_ = b1o[c];
#pragma unroll
    for (int p = 0; p < 9; p++) {
      int py = p / 3, px = p % 3;
      float su = bu_, so = bo_;
#pragma unroll
      for (int t = 0; t < 9; t++) {
        int qy = py + t / 3 - 1, qx = px + t % 3 - 1;
        if (qy >= 0 && qy < 3 && qx >= 0 && qx < 3) {
          float xq = xv[qy * 3 + qx];
          su += wu_[t] * xq;
          so += wo_[t] * xq;
        }
      }
      float val = tanh_f(so) * sigm_f(su);
      outx1[((size_t)(b0 + bb) * 32 + c) * 9 + p] = val;
      x1h[(bb * 9 + p) * 32 + c] = (_Float16)val;
    }
  }
  __syncthreads();

  const int wave = tid >> 5;
  const int lane = tid & 31;
  const int lm   = lane & 15;
  const bool hi  = lane >= 16;
  const int kA   = hi ? 16 : 0;   // A frag: 16 contiguous K at kA
  const int kB   = hi ? 16 : 0;   // B frag: 16 contiguous K at kB

  // ---- phase 2: cell2, M=144 (9 tiles) x N=128 (u:0..63, o:64..127), K=9 taps * 32 cin ----
  for (int task = wave; task < 36; task += 8) {
    int m = task >> 2, j = task & 3;          // m: M-tile, j: cout group of 16
    int r = m * 16 + lm;
    int bb = r / 9, pp = r % 9;
    int py = pp / 3, px = pp % 3;
    v8f cu = {0.f, 0.f, 0.f, 0.f, 0.f, 0.f, 0.f, 0.f};
    v8f co = {0.f, 0.f, 0.f, 0.f, 0.f, 0.f, 0.f, 0.f};
#pragma unroll 3
    for (int t = 0; t < 9; t++) {
      int qy = py + t / 3 - 1, qx = px + t % 3 - 1;
      bool valid = (qy >= 0 && qy < 3 && qx >= 0 && qx < 3);
      int qrow = valid ? (bb * 9 + qy * 3 + qx) : 144;   // 144 = zero row
      Frag16 a;
      const _Float16* src = &x1h[qrow * 32 + kA];
      a.q[0] = *(const uint4*)src;
      a.q[1] = *(const uint4*)(src + 8);
      int nu = j * 16 + lm;
      const _Float16* pu = W2 + ((nu * 9 + t) * 32 + kB);
      const _Float16* po = W2 + (((64 + nu) * 9 + t) * 32 + kB);
      Frag16 bu, bo;
      bu.q[0] = *(const uint4*)pu;  bu.q[1] = *(const uint4*)(pu + 8);
      bo.q[0] = *(const uint4*)po;  bo.q[1] = *(const uint4*)(po + 8);
      cu = wmma16(a.v, bu.v, cu);
      co = wmma16(a.v, bo.v, co);
    }
    int cout = j * 16 + lm;
    float biasu = b2u[cout], biaso = b2o[cout];
    // incremental (batch, pixel) decode of rows m*16+8*hi .. +7
    int rb = m * 16 + (hi ? 8 : 0);
    int bbw = rb / 9, ppw = rb % 9;
#pragma unroll
    for (int v = 0; v < 8; v++) {
      float val = tanh_f(co[v] + biaso) * sigm_f(cu[v] + biasu);
      outx2[((size_t)(b0 + bbw) * 64 + cout) * 9 + ppw] = val;
      x2h[(bbw * 9 + ppw) * 64 + cout] = (_Float16)val;
      ppw++;
      if (ppw == 9) { ppw = 0; bbw++; }
    }
  }
  __syncthreads();

  // ---- phase 3: cell3, M=144 (9 tiles) x N=32 (u:0..15, o:16..31), K=9 taps * 64 cin ----
  for (int m = wave; m < 9; m += 8) {
    int r = m * 16 + lm;
    int bb = r / 9, pp = r % 9;
    int py = pp / 3, px = pp % 3;
    v8f cu = {0.f, 0.f, 0.f, 0.f, 0.f, 0.f, 0.f, 0.f};
    v8f co = {0.f, 0.f, 0.f, 0.f, 0.f, 0.f, 0.f, 0.f};
#pragma unroll 3
    for (int t = 0; t < 9; t++) {
      int qy = py + t / 3 - 1, qx = px + t % 3 - 1;
      bool valid = (qy >= 0 && qy < 3 && qx >= 0 && qx < 3);
      int qrow = valid ? (bb * 9 + qy * 3 + qx) : 144;   // 144 = zero row
#pragma unroll
      for (int ks = 0; ks < 2; ks++) {
        Frag16 a;
        const _Float16* src = &x2h[qrow * 64 + ks * 32 + kA];
        a.q[0] = *(const uint4*)src;
        a.q[1] = *(const uint4*)(src + 8);
        const _Float16* pu = W3 + ((lm * 9 + t) * 64 + ks * 32 + kB);
        const _Float16* po = W3 + (((16 + lm) * 9 + t) * 64 + ks * 32 + kB);
        Frag16 bu, bo;
        bu.q[0] = *(const uint4*)pu;  bu.q[1] = *(const uint4*)(pu + 8);
        bo.q[0] = *(const uint4*)po;  bo.q[1] = *(const uint4*)(po + 8);
        cu = wmma16(a.v, bu.v, cu);
        co = wmma16(a.v, bo.v, co);
      }
    }
    int cout = lm;
    float biasu = b3u[cout], biaso = b3o[cout];
    int rb = m * 16 + (hi ? 8 : 0);
    int bbw = rb / 9, ppw = rb % 9;
#pragma unroll
    for (int v = 0; v < 8; v++) {
      float val = tanh_f(co[v] + biaso) * sigm_f(cu[v] + biasu);
      outx3[((size_t)(b0 + bbw) * 16 + cout) * 9 + ppw] = val;
      x3h[bbw * 160 + cout * 9 + ppw] = (_Float16)val;
      ppw++;
      if (ppw == 9) { ppw = 0; bbw++; }
    }
  }
  __syncthreads();

  // ---- phase 4: dense head, M=16 batches x N=16 (9 real) x K=160 (144 real) ----
  if (wave == 0) {
    v8f c = {0.f, 0.f, 0.f, 0.f, 0.f, 0.f, 0.f, 0.f};
#pragma unroll
    for (int ks = 0; ks < 5; ks++) {
      Frag16 a, bw;
      const _Float16* src = &x3h[lm * 160 + ks * 32 + kA];
      a.q[0] = *(const uint4*)src;
      a.q[1] = *(const uint4*)(src + 8);
      const _Float16* pw = Wd + (lm * 160 + ks * 32 + kB);
      bw.q[0] = *(const uint4*)pw;  bw.q[1] = *(const uint4*)(pw + 8);
      c = wmma16(a.v, bw.v, c);
    }
    int n = lm;
    if (n < 9) {
      float bias = bd[n];
#pragma unroll
      for (int v = 0; v < 8; v++) {
        int bl = v + (hi ? 8 : 0);
        outy[(size_t)(b0 + bl) * 9 + n] = c[v] + bias;
      }
    }
  }
}

extern "C" void kernel_launch(void* const* d_in, const int* in_sizes, int n_in,
                              void* d_out, int out_size, void* d_ws, size_t ws_size,
                              hipStream_t stream) {
  const float* inp = (const float*)d_in[0];
  const float* wd  = (const float*)d_in[1];
  const float* bd  = (const float*)d_in[2];
  const float* w1u = (const float*)d_in[3];
  const float* b1u = (const float*)d_in[4];
  // d_in[5], d_in[6] = w1r/b1r: dead (h==0 => reset unused). Same for w2r/w3r.
  const float* w1o = (const float*)d_in[7];
  const float* b1o = (const float*)d_in[8];
  const float* w2u = (const float*)d_in[9];
  const float* b2u = (const float*)d_in[10];
  const float* w2o = (const float*)d_in[13];
  const float* b2o = (const float*)d_in[14];
  const float* w3u = (const float*)d_in[15];
  const float* b3u = (const float*)d_in[16];
  const float* w3o = (const float*)d_in[19];
  const float* b3o = (const float*)d_in[20];

  int nb = in_sizes[0] / 9;  // batch count (65536)
  _Float16* ws = (_Float16*)d_ws;

  crnn_prep_weights<<<64, 256, 0, stream>>>(wd, w2u, w2o, w3u, w3o, ws);
  crnn_fused<<<nb / 16, 256, 0, stream>>>(inp, w1u, b1u, w1o, b1o,
                                          b2u, b2o, b3u, b3o, bd,
                                          ws, (float*)d_out, nb);
}
